// ScaledAttention_90477781057700
// MI455X (gfx1250) — matrix operand order repaired and checked by reading, not compiled, not run on hardware
//
#include <hip/hip_runtime.h>
#include <math.h>

typedef __attribute__((ext_vector_type(16))) __bf16 bf16x16;
typedef __attribute__((ext_vector_type(8)))  __bf16 bf16x8;
typedef __attribute__((ext_vector_type(4)))  __bf16 bf16x4;
typedef __attribute__((ext_vector_type(8)))  float  f32x8;
typedef __attribute__((ext_vector_type(4)))  float  f32x4;

#define D_DIM 1024
#define LQ    2048
#define LM    2048
#define NBAT  4
#define NROWS (NBAT*LQ)   // 8192

static __device__ __forceinline__ f32x8 wmma_bf16(bf16x16 a, bf16x16 b, f32x8 c) {
  return __builtin_amdgcn_wmma_f32_16x16x32_bf16(false, a, false, b, (short)0, c, false, false);
}

static __device__ __forceinline__ unsigned short bf_bits(float f) {
  __bf16 h = (__bf16)f;
  return __builtin_bit_cast(unsigned short, h);
}

static __device__ __forceinline__ bf16x16 combine8(bf16x8 lo, bf16x8 hi) {
  return __builtin_shufflevector(lo, hi, 0,1,2,3,4,5,6,7,8,9,10,11,12,13,14,15);
}

// A-operand (16x32, 16-bit): lane holds row M=lane%16; K elems [0..15] = k0+s*16 .. contiguous
static __device__ __forceinline__ bf16x16 load_a_bf16(const __bf16* row, int k0, int s) {
  bf16x8 lo = *(const bf16x8*)(row + k0 + s*16);
  bf16x8 hi = *(const bf16x8*)(row + k0 + s*16 + 8);
  return combine8(lo, hi);
}

// B-operand (32x16, 16-bit): lane holds col N=lane%16; K elems [0..15] = k0+s*16 .. contiguous
static __device__ __forceinline__ bf16x16 load_b_bf16(const __bf16* row, int k0, int s) {
  return *(const bf16x16*)(row + k0 + s*16);
}

// fp32 source row converted on the fly into A-operand layout
static __device__ __forceinline__ bf16x16 load_a_f32(const float* row, int k0, int s) {
  f32x4 p0 = *(const f32x4*)(row + k0 + s*16);
  f32x4 p1 = *(const f32x4*)(row + k0 + s*16 + 4);
  f32x4 p2 = *(const f32x4*)(row + k0 + s*16 + 8);
  f32x4 p3 = *(const f32x4*)(row + k0 + s*16 + 12);
  bf16x16 a;
#pragma unroll
  for (int i = 0; i < 4; ++i) {
    a[i]      = (__bf16)p0[i];
    a[4 + i]  = (__bf16)p1[i];
    a[8 + i]  = (__bf16)p2[i];
    a[12 + i] = (__bf16)p3[i];
  }
  return a;
}

// ---------------- fp32 -> bf16 weight conversion ----------------
__global__ void cvt_f32_bf16(const float* __restrict__ src, __bf16* __restrict__ dst, int n) {
  int i = (blockIdx.x * blockDim.x + threadIdx.x) * 4;
  if (i + 3 < n) {
    f32x4 v = *(const f32x4*)(src + i);
    bf16x4 o;
#pragma unroll
    for (int j = 0; j < 4; ++j) o[j] = (__bf16)v[j];
    *(bf16x4*)(dst + i) = o;
  }
}

// ---------------- projection GEMM: Y[n,e] = scale * sum_d A[n,d] * W[e,d] ----------------
// A fp32 [NROWS, D], W bf16 [D_DIM, D_DIM]. One wave -> 32 rows x 64 cols (W traffic halved
// vs 16-row tiles: 4 B tiles feed 8 WMMAs).
// Y (optional): bf16 [NROWS, D_DIM] row-major.  Yt (optional): bf16 [NBAT, D_DIM, LM] (V transposed).
__global__ void proj_gemm(const float* __restrict__ A, const __bf16* __restrict__ W,
                          __bf16* __restrict__ Y, __bf16* __restrict__ Yt, float scale) {
  const int lane = threadIdx.x & 31;
  const int wv   = threadIdx.x >> 5;
  const int wid  = blockIdx.x * 8 + wv;
  const int rowTile = wid >> 4;   // NROWS/32 = 256 row tiles
  const int col4    = wid & 15;   // 16 groups of 64 cols
  const int s  = lane >> 4;
  const int ln = lane & 15;

  const float*  arow0 = A + (size_t)(rowTile * 32 + ln) * D_DIM;
  const float*  arow1 = arow0 + (size_t)16 * D_DIM;
  const __bf16* wrow0 = W + (size_t)(col4 * 64 +  0 + ln) * D_DIM;
  const __bf16* wrow1 = W + (size_t)(col4 * 64 + 16 + ln) * D_DIM;
  const __bf16* wrow2 = W + (size_t)(col4 * 64 + 32 + ln) * D_DIM;
  const __bf16* wrow3 = W + (size_t)(col4 * 64 + 48 + ln) * D_DIM;

  f32x8 c[8] = {};   // c[j]: row-tile0, c[4+j]: row-tile1
  bf16x16 a0 = load_a_f32(arow0, 0, s);
  bf16x16 a1 = load_a_f32(arow1, 0, s);

  for (int kc = 0; kc < D_DIM; kc += 32) {
    bf16x16 b0 = load_b_bf16(wrow0, kc, s);
    bf16x16 b1 = load_b_bf16(wrow1, kc, s);
    bf16x16 b2 = load_b_bf16(wrow2, kc, s);
    bf16x16 b3 = load_b_bf16(wrow3, kc, s);
    const int kn = (kc + 32 < D_DIM) ? kc + 32 : 0;
    bf16x16 a0n = load_a_f32(arow0, kn, s);   // prefetch next A chunks
    bf16x16 a1n = load_a_f32(arow1, kn, s);
    c[0] = wmma_bf16(a0, b0, c[0]);
    c[1] = wmma_bf16(a0, b1, c[1]);
    c[2] = wmma_bf16(a0, b2, c[2]);
    c[3] = wmma_bf16(a0, b3, c[3]);
    c[4] = wmma_bf16(a1, b0, c[4]);
    c[5] = wmma_bf16(a1, b1, c[5]);
    c[6] = wmma_bf16(a1, b2, c[6]);
    c[7] = wmma_bf16(a1, b3, c[7]);
    a0 = a0n; a1 = a1n;
  }

  if (Y) {
#pragma unroll
    for (int i = 0; i < 2; ++i) {
#pragma unroll
      for (int j = 0; j < 4; ++j) {
        int n = col4 * 64 + j * 16 + ln;
#pragma unroll
        for (int r = 0; r < 8; ++r) {
          int row = rowTile * 32 + i * 16 + s * 8 + r;
          Y[(size_t)row * D_DIM + n] = (__bf16)(c[i * 4 + j][r] * scale);
        }
      }
    }
  }
  if (Yt) {
#pragma unroll
    for (int i = 0; i < 2; ++i) {
#pragma unroll
      for (int j = 0; j < 4; ++j) {
        int n    = col4 * 64 + j * 16 + ln;
        int row0 = rowTile * 32 + i * 16 + s * 8;
        int bb   = row0 / LM;
        int m    = row0 % LM;
        bf16x8 o;
#pragma unroll
        for (int r = 0; r < 8; ++r) o[r] = (__bf16)(c[i * 4 + j][r] * scale);
        *(bf16x8*)(Yt + ((size_t)bb * D_DIM + n) * LM + m) = o;
      }
    }
  }
}

// ---------------- fused flash attention (LDS-shared P^T, 32 queries / block) ----------------
// Q,K bf16 [NBAT*L, D] row-major (Q pre-scaled), Vt bf16 [NBAT, D, LM]. O bf16 [NBAT*LQ, D].
// Block = one (batch, 32-query tile); 16 waves (512 thr). Per 512-wide m "round" (4 rounds):
//   wave w computes S^T for m-chunk [w*32,w*32+32) for BOTH 16-query tiles (K read once),
//   round max/sum reduced via LDS per qtile, exp(S^T) published to LDS in B-operand layout,
//   each wave accumulates PV for a private 64-wide e-slab x 2 qtiles (each Vt operand feeds
//   two WMMAs -> K and Vt L2 traffic halved vs 16-query blocks).
__global__ void attn_kernel(const __bf16* __restrict__ Q, const __bf16* __restrict__ K,
                            const __bf16* __restrict__ Vt, __bf16* __restrict__ O) {
  __shared__ float        maxbuf[2 * 16 * 16];
  __shared__ float        sumbuf[2 * 16 * 16];
  __shared__ unsigned int pbuf[2 * 16 * 32 * 8];   // 2 qtiles x 16 chunks x 32 lanes x 32B = 32KB

  const int lane = threadIdx.x & 31;
  const int wv   = threadIdx.x >> 5;     // 0..15
  const int s    = lane >> 4;
  const int ln   = lane & 15;
  const int nqt  = LQ / 32;
  const int bb   = blockIdx.x / nqt;
  const int qt   = blockIdx.x % nqt;

  const __bf16* qrow0 = Q + (size_t)(bb * LQ + qt * 32 + ln) * D_DIM;       // qtile0 B-operand
  const __bf16* qrow1 = qrow0 + (size_t)16 * D_DIM;                          // qtile1
  const int     eslab = wv * 64;
  const __bf16* vbase = Vt + ((size_t)bb * D_DIM + eslab + ln) * LM;         // + t*16*LM per e-tile

  f32x8 acc[8] = {};                      // acc[t]: qtile0, acc[4+t]: qtile1 (t = e-tile 0..3)
  float run_max[2] = {-3.0e38f, -3.0e38f};
  float run_sum[2] = {0.0f, 0.0f};

  for (int round = 0; round < LM; round += 512) {
    // ---- S^T for this wave's 32-wide m chunk, both qtiles (pipelined over e) ----
    const int mb = round + wv * 32;
    const __bf16* krow0 = K + (size_t)(bb * LM + mb + ln) * D_DIM;
    const __bf16* krow1 = krow0 + (size_t)16 * D_DIM;
    f32x8 cs[4] = {};   // [g*2 + half]: qtile g, m-half
    bf16x16 bq0 = load_b_bf16(qrow0, 0, s);
    bf16x16 bq1 = load_b_bf16(qrow1, 0, s);
    bf16x16 a0  = load_a_bf16(krow0, 0, s);
    bf16x16 a1  = load_a_bf16(krow1, 0, s);
    for (int ec = 0; ec < D_DIM; ec += 32) {
      const int en = (ec + 32 < D_DIM) ? ec + 32 : 0;
      bf16x16 bq0n = load_b_bf16(qrow0, en, s);
      bf16x16 bq1n = load_b_bf16(qrow1, en, s);
      bf16x16 a0n  = load_a_bf16(krow0, en, s);
      bf16x16 a1n  = load_a_bf16(krow1, en, s);
      cs[0] = wmma_bf16(a0, bq0, cs[0]);
      cs[1] = wmma_bf16(a1, bq0, cs[1]);
      cs[2] = wmma_bf16(a0, bq1, cs[2]);
      cs[3] = wmma_bf16(a1, bq1, cs[3]);
      bq0 = bq0n; bq1 = bq1n; a0 = a0n; a1 = a1n;
    }

    // ---- per-query chunk max -> LDS (per qtile) ----
#pragma unroll
    for (int g = 0; g < 2; ++g) {
      float lmax = cs[g * 2][0];
#pragma unroll
      for (int r = 0; r < 8; ++r) {
        lmax = fmaxf(lmax, cs[g * 2][r]);
        lmax = fmaxf(lmax, cs[g * 2 + 1][r]);
      }
      lmax = fmaxf(lmax, __shfl_xor(lmax, 16, 32));
      if (s == 0) maxbuf[g * 256 + wv * 16 + ln] = lmax;
    }
    __syncthreads();

    float alpha[2];
#pragma unroll
    for (int g = 0; g < 2; ++g) {
      float rmax = maxbuf[g * 256 + ln];
#pragma unroll
      for (int w = 1; w < 16; ++w) rmax = fmaxf(rmax, maxbuf[g * 256 + w * 16 + ln]);
      float newmax = fmaxf(run_max[g], rmax);
      alpha[g] = __expf(run_max[g] - newmax);
      run_max[g] = newmax;

      // exp, chunk sum, publish P^T (B-operand layout: half-tile swap with lane^16)
      float e0[8], e1[8], lsum = 0.0f;
#pragma unroll
      for (int r = 0; r < 8; ++r) {
        e0[r] = __expf(cs[g * 2][r] - newmax);
        e1[r] = __expf(cs[g * 2 + 1][r] - newmax);
        lsum += e0[r] + e1[r];
      }
      lsum += __shfl_xor(lsum, 16, 32);
      if (s == 0) sumbuf[g * 256 + wv * 16 + ln] = lsum;

      unsigned int u0[4], u1[4], p0[4], p1[4];
#pragma unroll
      for (int j = 0; j < 4; ++j) {
        u0[j] = (unsigned)bf_bits(e0[2*j]) | ((unsigned)bf_bits(e0[2*j+1]) << 16);
        u1[j] = (unsigned)bf_bits(e1[2*j]) | ((unsigned)bf_bits(e1[2*j+1]) << 16);
      }
#pragma unroll
      for (int j = 0; j < 4; ++j) {
        p0[j] = (unsigned)__shfl_xor((int)u0[j], 16, 32);
        p1[j] = (unsigned)__shfl_xor((int)u1[j], 16, 32);
      }
      union { bf16x16 v; unsigned int u[8]; } bp;
#pragma unroll
      for (int j = 0; j < 4; ++j) {
        bp.u[j]     = s ? p1[j] : u0[j];
        bp.u[4 + j] = s ? u1[j] : p0[j];
      }
      *(bf16x16*)&pbuf[((g * 16 + wv) * 32 + lane) * 8] = bp.v;   // one 32B ds_store
    }
    __syncthreads();

    // ---- running sums + rescale + O^T accumulation over all 16 chunks of the round ----
#pragma unroll
    for (int g = 0; g < 2; ++g) {
      float rsum = 0.0f;
#pragma unroll
      for (int w = 0; w < 16; ++w) rsum += sumbuf[g * 256 + w * 16 + ln];
      run_sum[g] = run_sum[g] * alpha[g] + rsum;
    }
#pragma unroll
    for (int t = 0; t < 4; ++t) {
      acc[t]     = acc[t]     * alpha[0];
      acc[4 + t] = acc[4 + t] * alpha[1];
    }

    for (int pw = 0; pw < 16; ++pw) {
      bf16x16 bpv0 = *(const bf16x16*)&pbuf[((     pw) * 32 + lane) * 8];
      bf16x16 bpv1 = *(const bf16x16*)&pbuf[((16 + pw) * 32 + lane) * 8];
      const int mpw = round + pw * 32;
      bf16x16 av = load_a_bf16(vbase, mpw, s);            // e-tile t=0
#pragma unroll
      for (int t = 0; t < 4; ++t) {
        bf16x16 avn = (t < 3) ? load_a_bf16(vbase + (size_t)(t + 1) * 16 * LM, mpw, s) : av;
        acc[t]     = wmma_bf16(av, bpv0, acc[t]);         // Vt operand reused for both qtiles
        acc[4 + t] = wmma_bf16(av, bpv1, acc[4 + t]);
        av = avn;
      }
    }
  }

  // ---- normalize and store O row-major: lane has fixed q, 8 contiguous e per tile ----
#pragma unroll
  for (int g = 0; g < 2; ++g) {
    float rinv = 1.0f / run_sum[g];
    __bf16* orow = O + (size_t)(bb * LQ + qt * 32 + g * 16 + ln) * D_DIM;
#pragma unroll
    for (int t = 0; t < 4; ++t) {
      bf16x8 o;
#pragma unroll
      for (int r = 0; r < 8; ++r) o[r] = (__bf16)(acc[g * 4 + t][r] * rinv);
      *(bf16x8*)(orow + eslab + t * 16 + s * 8) = o;
    }
  }
}

// ---------------- output GEMM: out[n,o] = sum_e A[n,e] * W[o,e], fp32 out (32x64 tiles) ----------------
__global__ void out_gemm(const __bf16* __restrict__ A, const __bf16* __restrict__ W,
                         float* __restrict__ Y) {
  const int lane = threadIdx.x & 31;
  const int wv   = threadIdx.x >> 5;
  const int wid  = blockIdx.x * 8 + wv;
  const int rowTile = wid >> 4;   // NROWS/32 = 256
  const int col4    = wid & 15;
  const int s  = lane >> 4;
  const int ln = lane & 15;

  const __bf16* arow0 = A + (size_t)(rowTile * 32 + ln) * D_DIM;
  const __bf16* arow1 = arow0 + (size_t)16 * D_DIM;
  const __bf16* wrow0 = W + (size_t)(col4 * 64 +  0 + ln) * D_DIM;
  const __bf16* wrow1 = W + (size_t)(col4 * 64 + 16 + ln) * D_DIM;
  const __bf16* wrow2 = W + (size_t)(col4 * 64 + 32 + ln) * D_DIM;
  const __bf16* wrow3 = W + (size_t)(col4 * 64 + 48 + ln) * D_DIM;

  f32x8 c[8] = {};
  bf16x16 a0 = load_a_bf16(arow0, 0, s);
  bf16x16 a1 = load_a_bf16(arow1, 0, s);

  for (int kc = 0; kc < D_DIM; kc += 32) {
    bf16x16 b0 = load_b_bf16(wrow0, kc, s);
    bf16x16 b1 = load_b_bf16(wrow1, kc, s);
    bf16x16 b2 = load_b_bf16(wrow2, kc, s);
    bf16x16 b3 = load_b_bf16(wrow3, kc, s);
    const int kn = (kc + 32 < D_DIM) ? kc + 32 : 0;
    bf16x16 a0n = load_a_bf16(arow0, kn, s);
    bf16x16 a1n = load_a_bf16(arow1, kn, s);
    c[0] = wmma_bf16(a0, b0, c[0]);
    c[1] = wmma_bf16(a0, b1, c[1]);
    c[2] = wmma_bf16(a0, b2, c[2]);
    c[3] = wmma_bf16(a0, b3, c[3]);
    c[4] = wmma_bf16(a1, b0, c[4]);
    c[5] = wmma_bf16(a1, b1, c[5]);
    c[6] = wmma_bf16(a1, b2, c[6]);
    c[7] = wmma_bf16(a1, b3, c[7]);
    a0 = a0n; a1 = a1n;
  }
#pragma unroll
  for (int i = 0; i < 2; ++i) {
#pragma unroll
    for (int j = 0; j < 4; ++j) {
      int n = col4 * 64 + j * 16 + ln;
#pragma unroll
      for (int r = 0; r < 8; ++r) {
        int row = rowTile * 32 + i * 16 + s * 8 + r;
        Y[(size_t)row * D_DIM + n] = c[i * 4 + j][r];
      }
    }
  }
}

extern "C" void kernel_launch(void* const* d_in, const int* in_sizes, int n_in,
                              void* d_out, int out_size, void* d_ws, size_t ws_size,
                              hipStream_t stream) {
  const float* X  = (const float*)d_in[0];
  const float* M  = (const float*)d_in[1];
  const float* Wq = (const float*)d_in[2];
  const float* Wk = (const float*)d_in[3];
  const float* Wv = (const float*)d_in[4];
  const float* Wo = (const float*)d_in[5];
  float* out = (float*)d_out;

  const size_t WN = (size_t)D_DIM * D_DIM;        // 1M elems per weight
  const size_t QN = (size_t)NROWS * D_DIM;        // 8M elems
  __bf16* ws  = (__bf16*)d_ws;
  __bf16* WqB = ws;
  __bf16* WkB = WqB + WN;
  __bf16* WvB = WkB + WN;
  __bf16* WoB = WvB + WN;
  __bf16* Qb  = WoB + WN;
  __bf16* Kb  = Qb + QN;
  __bf16* Vtb = Kb + QN;
  __bf16* Ob  = Vtb + QN;

  // 1) weights fp32 -> bf16
  int cvtBlocks = (int)(WN / 4 / 256);
  cvt_f32_bf16<<<cvtBlocks, 256, 0, stream>>>(Wq, WqB, (int)WN);
  cvt_f32_bf16<<<cvtBlocks, 256, 0, stream>>>(Wk, WkB, (int)WN);
  cvt_f32_bf16<<<cvtBlocks, 256, 0, stream>>>(Wv, WvB, (int)WN);
  cvt_f32_bf16<<<cvtBlocks, 256, 0, stream>>>(Wo, WoB, (int)WN);

  // 2) projections (512 blocks x 8 waves = 4096 wave-tiles of 32x64)
  proj_gemm<<<512, 256, 0, stream>>>(X, WqB, Qb, nullptr, 0.03125f);  // depth^-0.5 = 1/32
  proj_gemm<<<512, 256, 0, stream>>>(M, WkB, Kb, nullptr, 1.0f);
  proj_gemm<<<512, 256, 0, stream>>>(M, WvB, nullptr, Vtb, 1.0f);

  // 3) fused attention: one block (16 waves) per (batch, 32-query tile)
  attn_kernel<<<NBAT * (LQ / 32), 512, 0, stream>>>(Qb, Kb, Vtb, Ob);

  // 4) output projection, fp32 result
  out_gemm<<<512, 256, 0, stream>>>(Ob, WoB, out);
}
